// SimCLRLoss_90718299226144
// MI455X (gfx1250) — matrix operand order repaired and checked by reading, not compiled, not run on hardware
//
#include <hip/hip_runtime.h>
#include <hip/hip_bf16.h>

typedef __attribute__((ext_vector_type(16))) _Float16 v16h;
typedef __attribute__((ext_vector_type(8)))  _Float16 v8h;
typedef __attribute__((ext_vector_type(8)))  float    v8f;
typedef __attribute__((ext_vector_type(4)))  float    v4f;

#define BATCH   4096
#define EMBED   128
#define ROWS    (2 * BATCH)     // 8192
#define INV_T   2.0f            // 1 / TEMPERATURE
#define EPS_N   1e-8f
// exp(x/T) = 2^(x * (1/T)*log2(e));  (1/T)*log2(e) = 2 * 1.4426950408889634
#define EXP2_SCALE 2.8853900817779268f

__device__ __forceinline__ float wave_reduce_add32(float v) {
    #pragma unroll
    for (int m = 1; m < 32; m <<= 1) v += __shfl_xor(v, m, 32);
    return v;
}

// -----------------------------------------------------------------------------
// Kernel 1: L2-normalize both projections (f32 in -> f16 out, L2-resident z),
// and compute pos[i] = dot(z_i, z_j) = dot(p1,p2)/(|p1||p2|).
// One wave32 per row pair; each lane handles 4 contiguous floats (float4 load).
// -----------------------------------------------------------------------------
__global__ __launch_bounds__(256) void simclr_normalize(
        const float* __restrict__ p1, const float* __restrict__ p2,
        _Float16* __restrict__ zh, float* __restrict__ pos) {
    const int pair = (blockIdx.x * blockDim.x + threadIdx.x) >> 5;  // row index
    const int lane = threadIdx.x & 31;
    if (pair >= BATCH) return;

    const v4f a = *(const v4f*)(p1 + (size_t)pair * EMBED + lane * 4);
    const v4f b = *(const v4f*)(p2 + (size_t)pair * EMBED + lane * 4);

    float ss1 = 0.f, ss2 = 0.f, dp = 0.f;
    #pragma unroll
    for (int c = 0; c < 4; ++c) {
        ss1 += a[c] * a[c];
        ss2 += b[c] * b[c];
        dp  += a[c] * b[c];
    }
    ss1 = wave_reduce_add32(ss1);
    ss2 = wave_reduce_add32(ss2);
    dp  = wave_reduce_add32(dp);

    const float i1 = 1.0f / fmaxf(sqrtf(ss1), EPS_N);
    const float i2 = 1.0f / fmaxf(sqrtf(ss2), EPS_N);

    _Float16* z1 = zh + (size_t)pair * EMBED + lane * 4;
    _Float16* z2 = zh + (size_t)(pair + BATCH) * EMBED + lane * 4;
    #pragma unroll
    for (int c = 0; c < 4; ++c) {
        z1[c] = (_Float16)(a[c] * i1);
        z2[c] = (_Float16)(b[c] * i2);
    }
    if (lane == 0) pos[pair] = dp * (i1 * i2);
}

// -----------------------------------------------------------------------------
// Kernel 2: fused  sim = z·zT  +  rowwise sum(exp(sim/T)), diagonal removed.
// One block per 16-row tile (512 blocks); 8 waves, each wave owns a 1024-col
// stripe (64 column tiles). K=128 -> 4 x v_wmma_f32_16x16x32_f16 per tile.
// B fragments are double-buffered in registers: the load clause for tile t+1
// issues before tile t's WMMA chain, and each WMMA waits only for its own
// fragment (partial s_wait_loadcnt). The exp epilogue (raw v_exp_f32 via
// __builtin_amdgcn_exp2f with the 1/T*log2e scale pre-folded) of tile t
// co-executes with tile t+1's XDL WMMAs.
//
// Diagonal handling: accumulate everything, then subtract exp(|z_r|^2 / T)
// once per row; the self-dot is reconstructed from the register-resident A
// fragments (lane L and L+16 hold complementary K halves of row L%16).
//
// CDNA5 16-bit A layout (16x32): lane L and L+16 both hold row M=L%16;
//   as loaded here: lo lanes K = k0..k0+15, hi lanes K = k0+16..k0+31 (same k order as B).
// B layout (32x16): lane L holds column N=L%16, K = k0 + (L>=16)*16 + {0..15}
//   -> one contiguous 32-byte load of a z row.
// C/D layout: VGPR r, lane L -> M = r + 8*(L>=16), N = L%16.
// -----------------------------------------------------------------------------
__global__ __launch_bounds__(256) void simclr_simsum(
        const _Float16* __restrict__ zh, float* __restrict__ rowsum) {
    __shared__ float smem[16];
    __shared__ float sdiag[16];
    const int lane = threadIdx.x & 31;
    const int wid  = threadIdx.x >> 5;        // 0..7
    const int hi   = lane >> 4;               // lane-half select
    const int ln   = lane & 15;
    const int rowbase = blockIdx.x << 4;      // 16-row output tile

    if (threadIdx.x < 16) smem[threadIdx.x] = 0.f;
    __syncthreads();

    // ---- A fragments: loaded once, reused for all 64 column tiles ----
    v16h afrag[4];
    {
        const _Float16* arow = zh + (size_t)(rowbase + ln) * EMBED;
        #pragma unroll
        for (int kc = 0; kc < 4; ++kc) {
            const int k0 = kc * 32 + hi * 16;
            const v8h lo8 = *(const v8h*)(arow + k0);
            const v8h hi8 = *(const v8h*)(arow + k0 + 8);
            #pragma unroll
            for (int j = 0; j < 8; ++j) {
                afrag[kc][j]     = lo8[j];
                afrag[kc][j + 8] = hi8[j];
            }
        }
    }

    // ---- Diagonal term exp(|z_r|^2 / T), computed once by wave 0 ----
    if (wid == 0) {
        float d = 0.f;
        #pragma unroll
        for (int kc = 0; kc < 4; ++kc)
            #pragma unroll
            for (int j = 0; j < 16; ++j) {
                const float x = (float)afrag[kc][j];
                d += x * x;
            }
        d += __shfl_xor(d, 16, 32);           // combine the two K-halves
        if (lane < 16) sdiag[lane] = __builtin_amdgcn_exp2f(d * EXP2_SCALE);
    }

    float rs[8];
    #pragma unroll
    for (int r = 0; r < 8; ++r) rs[r] = 0.f;

    constexpr int NT = (ROWS / 8) / 16;       // 64 column tiles per wave
    const _Float16* bptr =
        zh + (size_t)(wid * (ROWS / 8) + ln) * EMBED + hi * 16;
    constexpr size_t TILE_STRIDE = (size_t)16 * EMBED;   // halves per col tile

    v16h b0[4], b1[4];
    #pragma unroll
    for (int kc = 0; kc < 4; ++kc) b0[kc] = *(const v16h*)(bptr + kc * 32);

    #pragma unroll 1
    for (int t = 0; t < NT; t += 2) {
        // prefetch tile t+1 into the other buffer
        const _Float16* p1b = bptr + TILE_STRIDE;
        #pragma unroll
        for (int kc = 0; kc < 4; ++kc) b1[kc] = *(const v16h*)(p1b + kc * 32);

        {   // process tile t from b0
            v8f acc = {};
            #pragma unroll
            for (int kc = 0; kc < 4; ++kc)
                acc = __builtin_amdgcn_wmma_f32_16x16x32_f16(
                          false, afrag[kc], false, b0[kc],
                          (short)0, acc, false, false);
            #pragma unroll
            for (int r = 0; r < 8; ++r)
                rs[r] += __builtin_amdgcn_exp2f(acc[r] * EXP2_SCALE);
        }

        // Unguarded prefetch of tile t+2 into b0: on the final iteration this
        // over-reads one 4KB tile past the z region, which lands in the
        // pos/rowsum area of the same workspace allocation (valid memory,
        // values never consumed) -- removes the per-iteration branch.
        {
            const _Float16* p2b = bptr + 2 * TILE_STRIDE;
            #pragma unroll
            for (int kc = 0; kc < 4; ++kc) b0[kc] = *(const v16h*)(p2b + kc * 32);
        }

        {   // process tile t+1 from b1
            v8f acc = {};
            #pragma unroll
            for (int kc = 0; kc < 4; ++kc)
                acc = __builtin_amdgcn_wmma_f32_16x16x32_f16(
                          false, afrag[kc], false, b1[kc],
                          (short)0, acc, false, false);
            #pragma unroll
            for (int r = 0; r < 8; ++r)
                rs[r] += __builtin_amdgcn_exp2f(acc[r] * EXP2_SCALE);
        }

        bptr += 2 * TILE_STRIDE;
    }

    // Reduce over the 16 lanes (N) inside each lane-half: xor masks < 16
    // stay within the half, so all lanes of a half end with the row total.
    #pragma unroll
    for (int r = 0; r < 8; ++r) {
        #pragma unroll
        for (int m = 1; m < 16; m <<= 1) rs[r] += __shfl_xor(rs[r], m, 32);
    }

    if (ln == 0) {
        #pragma unroll
        for (int r = 0; r < 8; ++r) atomicAdd(&smem[r + hi * 8], rs[r]);
    }
    __syncthreads();

    if (threadIdx.x < 16)
        rowsum[rowbase + threadIdx.x] = smem[threadIdx.x] - sdiag[threadIdx.x];
}

// -----------------------------------------------------------------------------
// Kernel 3: loss = (1/2B) * sum_r [ log(rowsum[r]) - pos[r % B] / T ]
// -----------------------------------------------------------------------------
__global__ __launch_bounds__(1024) void simclr_loss(
        const float* __restrict__ rowsum, const float* __restrict__ pos,
        float* __restrict__ out) {
    __shared__ float smem[32];
    float v = 0.f;
    for (int r = threadIdx.x; r < ROWS; r += 1024)
        v += __logf(rowsum[r]) - pos[r & (BATCH - 1)] * INV_T;

    v = wave_reduce_add32(v);
    const int lane = threadIdx.x & 31;
    const int wid  = threadIdx.x >> 5;
    if (lane == 0) smem[wid] = v;
    __syncthreads();
    if (wid == 0) {
        float t = smem[lane];                  // 32 waves -> smem[32]
        t = wave_reduce_add32(t);
        if (lane == 0) out[0] = t / (float)ROWS;
    }
}

// -----------------------------------------------------------------------------
extern "C" void kernel_launch(void* const* d_in, const int* in_sizes, int n_in,
                              void* d_out, int out_size, void* d_ws, size_t ws_size,
                              hipStream_t stream) {
    const float* p1 = (const float*)d_in[0];
    const float* p2 = (const float*)d_in[1];

    // Workspace carve-up: z(f16) 2MB | pos 16KB | rowsum 32KB
    // (pos/rowsum also absorb the benign one-tile prefetch over-read)
    _Float16* zh     = (_Float16*)d_ws;
    float*    pos    = (float*)((char*)d_ws + (size_t)ROWS * EMBED * sizeof(_Float16));
    float*    rowsum = pos + BATCH;

    simclr_normalize<<<BATCH / 8, 256, 0, stream>>>(p1, p2, zh, pos);
    simclr_simsum<<<ROWS / 16, 256, 0, stream>>>(zh, rowsum);
    simclr_loss<<<1, 1024, 0, stream>>>(rowsum, pos, (float*)d_out);
}
